// RNN_16681652978222
// MI455X (gfx1250) — matrix operand order repaired and checked by reading, not compiled, not run on hardware
//
#include <hip/hip_runtime.h>
#include <math.h>

// ---------------------------------------------------------------------------
// RNN on MI455X (gfx1250): batch-parallel recurrence, bf16 WMMA, async LDS.
//   B=256 T=512 D_IN=128 H=512 D_OUT=64
// Pass 1: transpose weights -> bf16 column-major (contiguous B-fragments)
// Pass 2: xproj[t][b][h] = x[b][t][:]@Win + brec   (bf16, [T,B,H])
// Pass 3: 16 workgroups x 1024 thr; WG owns 16 batch rows, wave w owns 16 h
//         columns. Per step: C=xp tile (async-prefetched to LDS, depth 4),
//         16x v_wmma_f32_16x16x32_bf16 over K=512 with A=h from LDS,
//         tanh, ping-pong h in LDS, spill h to global history.
// Pass 4: out = h_hist @ Wout + bout  (parallel WMMA GEMM)
// ---------------------------------------------------------------------------

typedef __bf16 bf16;
typedef __attribute__((ext_vector_type(16))) __bf16 v16bf;
typedef __attribute__((ext_vector_type(8)))  float  v8f;
typedef __attribute__((ext_vector_type(4)))  float  v4f;
typedef __attribute__((ext_vector_type(4)))  int    v4i;
typedef __attribute__((ext_vector_type(8)))  int    v8i;

#define B_   256
#define T_   512
#define DIN  128
#define H_   512
#define DOUT 64
#define HPAD 520   // LDS row pitch in bf16: 1040B -> stride 4 banks mod 64
#define XPBUFS 4   // xp prefetch pipeline depth

#if __has_builtin(__builtin_amdgcn_global_load_async_to_lds_b128)
#define USE_ASYNC_LDS 1
#endif

union AFrag { v16bf bf; v4i d4[2]; };
union BFrag { v16bf bf; v8i d8; };

__device__ inline v8f wmma_bf16(v16bf a, v16bf b, v8f c) {
  return __builtin_amdgcn_wmma_f32_16x16x32_bf16(
      false, a, false, b, (short)0, c, false, false);
}

__device__ inline float fast_tanh(float x) {
#if __has_builtin(__builtin_amdgcn_tanhf)
  return __builtin_amdgcn_tanhf(x);
#elif __has_builtin(__builtin_amdgcn_tanh_f32)
  return __builtin_amdgcn_tanh_f32(x);
#else
  return tanhf(x);
#endif
}

// copy 16 bytes: global -> LDS (async on CDNA5, sync fallback otherwise).
// Probe-learned signature: (v4i as(1)* src, v4i as(3)* dst, Imm off, Imm cpol)
__device__ inline void cp16_g2lds(bf16* dst_lds, const bf16* src) {
#ifdef USE_ASYNC_LDS
  __builtin_amdgcn_global_load_async_to_lds_b128(
      (__attribute__((address_space(1))) v4i*)src,
      (__attribute__((address_space(3))) v4i*)dst_lds, 0, 0);
#else
  *(v4i*)dst_lds = *(const v4i*)src;
#endif
}

template <int N> __device__ inline void wait_async_le() {
#ifdef USE_ASYNC_LDS
#if __has_builtin(__builtin_amdgcn_s_wait_asynccnt)
  __builtin_amdgcn_s_wait_asynccnt(N);
#else
  asm volatile("s_wait_asynccnt %0" ::"i"(N) : "memory");
#endif
#endif
}

// ---------------- Pass 1: transpose f32 [R][C] -> bf16 [C][R] ---------------
__global__ void k_transpose_bf16(const float* __restrict__ src,
                                 bf16* __restrict__ dst, int R, int C) {
  int i = blockIdx.x * blockDim.x + threadIdx.x;
  if (i >= R * C) return;
  int k = i / C, n = i % C;
  dst[(size_t)n * R + k] = (bf16)src[(size_t)k * C + n];
}

// ---------------- Pass 2: xproj = x@Win + brec  -> bf16 [T][B][H] -----------
__global__ __launch_bounds__(256) void k_xproj(
    const float* __restrict__ x,     // [B][T][DIN]
    const bf16* __restrict__ WinT,   // [H][DIN]
    const float* __restrict__ brec,  // [H]
    bf16* __restrict__ xp) {         // [T][B][H]
  const int lane = threadIdx.x & 31;
  const int wid  = (blockIdx.x * blockDim.x + threadIdx.x) >> 5;
  const int rt = wid >> 5, ct = wid & 31;     // 8192 row tiles x 32 col tiles
  const int t = rt >> 4, b0 = (rt & 15) << 4, n0 = ct << 4;
  const int r = lane & 15, half = lane >> 4;

  v8f c;
  const float ci = brec[n0 + r];
#pragma unroll
  for (int v = 0; v < 8; v++) c[v] = ci;

  const bf16*  bp = WinT + (size_t)(n0 + r) * DIN;
  const float* ap = x + ((size_t)(b0 + r) * T_ + t) * DIN;

#pragma unroll
  for (int kc = 0; kc < DIN / 32; ++kc) {
    const int kb = kc * 32;
    AFrag b; b.d4[0] = *(const v4i*)(bp + kb + 8 * half); b.d4[1] = *(const v4i*)(bp + kb + 16 + 8 * half);
    v4f f0a = *(const v4f*)(ap + kb + 8 * half);
    v4f f0b = *(const v4f*)(ap + kb + 8 * half + 4);
    v4f f1a = *(const v4f*)(ap + kb + 16 + 8 * half);
    v4f f1b = *(const v4f*)(ap + kb + 16 + 8 * half + 4);
    AFrag a;
#pragma unroll
    for (int e = 0; e < 4; e++) {
      a.bf[e]      = (bf16)f0a[e];  a.bf[4 + e]  = (bf16)f0b[e];
      a.bf[8 + e]  = (bf16)f1a[e];  a.bf[12 + e] = (bf16)f1b[e];
    }
    c = wmma_bf16(a.bf, b.bf, c);
  }
  bf16* orow = xp + ((size_t)t * B_ + b0) * H_ + n0 + r;
#pragma unroll
  for (int v = 0; v < 8; v++) orow[(size_t)(v + 8 * half) * H_] = (bf16)c[v];
}

// ---------------- Pass 3: sequential recurrence -----------------------------
__global__ __launch_bounds__(1024) void k_rnn(
    const bf16* __restrict__ xp,     // [T][B][H]
    const bf16* __restrict__ WrecT,  // [H(n)][H(k)]
    const float* __restrict__ h0,    // [H]
    bf16* __restrict__ hhist) {      // [T][B][H]
  __shared__ bf16 hbuf[2][16][HPAD];
  __shared__ bf16 xpb[XPBUFS][16 * HPAD];

  const int tid  = threadIdx.x;
  const int lane = tid & 31;
  const int w    = tid >> 5;                 // wave id 0..31 -> n slice
  const int b0   = blockIdx.x << 4;
  const int r = lane & 15, half = lane >> 4;
  const int n = (w << 4) + r;                // this lane's h column

  // warm WGP$/L2 with this wave's Wrec slice (constant across all 512 steps)
  const bf16* wp = WrecT + (size_t)n * H_;
#pragma unroll
  for (int kc = 0; kc < 16; kc++)
    __builtin_prefetch(wp + kc * 32 + 16 * half, 0, 3);

  // h_0 broadcast: 1024 threads x 8 entries cover 16x512
  {
    int base = tid * 8;
    int m = base >> 9, col = base & 511;
#pragma unroll
    for (int e = 0; e < 8; e++) hbuf[0][m][col + e] = (bf16)h0[col + e];
  }

  // xp tile copy geometry: thread tid moves 16B; LDS rows padded to HPAD
  const bf16* xsrc = xp + (size_t)b0 * H_;
  const int xrow = tid >> 6, xcol = (tid & 63) * 8;

#pragma unroll
  for (int p = 0; p < XPBUFS - 1; ++p)
    cp16_g2lds(&xpb[p][xrow * HPAD + xcol],
               xsrc + (size_t)p * B_ * H_ + xrow * H_ + xcol);

  for (int t = 0; t < T_; ++t) {
    wait_async_le<XPBUFS - 2>();   // xp tile t has landed
    __syncthreads();               // h[t&1] + xp tile visible; old reads done
    if (t + XPBUFS - 1 < T_) {     // refill the buffer freed at step t-1
      const int tf = t + XPBUFS - 1;
      cp16_g2lds(&xpb[tf & (XPBUFS - 1)][xrow * HPAD + xcol],
                 xsrc + (size_t)tf * B_ * H_ + xrow * H_ + xcol);
    }
    const int cur = t & 1, nxt = cur ^ 1;

    // C init = xproj tile
    const bf16* xt = &xpb[t & (XPBUFS - 1)][0];
    v8f c;
#pragma unroll
    for (int v = 0; v < 8; v++) c[v] = (float)xt[(v + 8 * half) * HPAD + n];

    // h @ Wrec over K=512: A from LDS (ds_load_b128 x2), B from L2-resident
#pragma unroll
    for (int kc = 0; kc < 16; kc++) {
      AFrag a;
      const bf16* hr = &hbuf[cur][r][kc * 32 + 8 * half];
      a.d4[0] = *(const v4i*)hr;
      a.d4[1] = *(const v4i*)(hr + 16);
      AFrag b; b.d4[0] = *(const v4i*)(wp + kc * 32 + 8 * half); b.d4[1] = *(const v4i*)(wp + kc * 32 + 16 + 8 * half);
      c = wmma_bf16(a.bf, b.bf, c);
    }

    // tanh, ping-pong h in LDS, spill to global history (fire-and-forget)
    bf16* hd = hhist + ((size_t)t * B_ + b0) * H_ + n;
#pragma unroll
    for (int v = 0; v < 8; v++) {
      const int m = v + 8 * half;
      bf16 hb = (bf16)fast_tanh(c[v]);
      hbuf[nxt][m][n] = hb;
      hd[(size_t)m * H_] = hb;
    }
  }
}

// ---------------- Pass 4: out = h_hist @ Wout + bout ------------------------
__global__ __launch_bounds__(256) void k_out(
    const bf16* __restrict__ hhist,  // [T][B][H]
    const bf16* __restrict__ WoutT,  // [DOUT][H]
    const float* __restrict__ bout,  // [DOUT]
    float* __restrict__ out) {       // [B][T][DOUT]
  const int lane = threadIdx.x & 31;
  const int wid  = (blockIdx.x * blockDim.x + threadIdx.x) >> 5;
  const int rt = wid >> 2, ct = wid & 3;     // 8192 row tiles x 4 col tiles
  const int t = rt >> 4, b0 = (rt & 15) << 4, n0 = ct << 4;
  const int r = lane & 15, half = lane >> 4;

  v8f c;
  const float ci = bout[n0 + r];
#pragma unroll
  for (int v = 0; v < 8; v++) c[v] = ci;

  const bf16* ap = hhist + ((size_t)t * B_ + b0 + r) * H_;
  const bf16* bp = WoutT + (size_t)(n0 + r) * H_;
#pragma unroll
  for (int kc = 0; kc < 16; kc++) {
    AFrag a;
    a.d4[0] = *(const v4i*)(ap + kc * 32 + 8 * half);
    a.d4[1] = *(const v4i*)(ap + kc * 32 + 16 + 8 * half);
    AFrag b; b.d4[0] = *(const v4i*)(bp + kc * 32 + 8 * half); b.d4[1] = *(const v4i*)(bp + kc * 32 + 16 + 8 * half);
    c = wmma_bf16(a.bf, b.bf, c);
  }
  float* orow = out + ((size_t)b0 * T_ + t) * DOUT + n0 + r;
#pragma unroll
  for (int v = 0; v < 8; v++)
    orow[(size_t)(v + 8 * half) * T_ * DOUT] = c[v];
}

// ---------------------------------------------------------------------------
extern "C" void kernel_launch(void* const* d_in, const int* in_sizes, int n_in,
                              void* d_out, int out_size, void* d_ws,
                              size_t ws_size, hipStream_t stream) {
  (void)in_sizes; (void)n_in; (void)out_size; (void)ws_size;
  const float* x    = (const float*)d_in[0];
  const float* Win  = (const float*)d_in[1];
  const float* Wrec = (const float*)d_in[2];
  const float* brec = (const float*)d_in[3];
  const float* Wout = (const float*)d_in[4];
  const float* bout = (const float*)d_in[5];
  const float* h0   = (const float*)d_in[6];
  float* out = (float*)d_out;

  char* ws = (char*)d_ws;
  size_t off = 0;
  auto alloc = [&](size_t bytes) -> void* {
    off = (off + 255) & ~(size_t)255;
    void* p = ws + off;
    off += bytes;
    return p;
  };
  bf16* xpj   = (bf16*)alloc((size_t)T_ * B_ * H_ * sizeof(bf16));   // 134 MB
  bf16* hh    = (bf16*)alloc((size_t)T_ * B_ * H_ * sizeof(bf16));   // 134 MB
  bf16* WinT  = (bf16*)alloc((size_t)H_ * DIN * sizeof(bf16));
  bf16* WrecT = (bf16*)alloc((size_t)H_ * H_ * sizeof(bf16));
  bf16* WoutT = (bf16*)alloc((size_t)DOUT * H_ * sizeof(bf16));

  k_transpose_bf16<<<(DIN * H_) / 256, 256, 0, stream>>>(Win, WinT, DIN, H_);
  k_transpose_bf16<<<(H_ * H_) / 256, 256, 0, stream>>>(Wrec, WrecT, H_, H_);
  k_transpose_bf16<<<(H_ * DOUT) / 256, 256, 0, stream>>>(Wout, WoutT, H_, DOUT);
  k_xproj<<<(T_ * B_ / 16) * 32 / 8, 256, 0, stream>>>(x, WinT, brec, xpj);
  k_rnn<<<B_ / 16, 1024, 0, stream>>>(xpj, WrecT, h0, hh);
  k_out<<<(T_ * B_ / 16) * 4 / 8, 256, 0, stream>>>(hh, WoutT, bout, out);
}
